// BiLSTM_CRF_87892210745516
// MI455X (gfx1250) — compile-verified, matrix operand order checked by reading, not run on hardware
//
#include <hip/hip_runtime.h>
#include <hip/hip_bf16.h>

// ---------------------------------------------------------------------------
// Types for CDNA5 WMMA / TDM builtins
// ---------------------------------------------------------------------------
typedef __bf16 bf16_t;
typedef __bf16 v16bf __attribute__((ext_vector_type(16)));
typedef float  v8f   __attribute__((ext_vector_type(8)));
typedef unsigned int uint32;
typedef uint32 uint4v __attribute__((ext_vector_type(4)));
typedef int    int8v  __attribute__((ext_vector_type(8)));
typedef int    int4v  __attribute__((ext_vector_type(4)));

#define T_LEN 4096
#define E_DIM 1024
#define HH    512
#define G4    2048      // 4*HH
#define KTAG  16
#define START_TAG 14
#define STOP_TAG  15
#define NEG_F (-10000.0f)

// ---------------------------------------------------------------------------
// Tensor Data Mover: 2D tile load (bf16, data_size=2B) global -> LDS.
// Descriptor packing per CDNA5 ISA ch.8 (D# group0/group1).
// 6-arg builtin: (uint32x4 g0, int32x8 g1, int32x4, int32x4, int32x8, i32 cpol)
// ---------------------------------------------------------------------------
__device__ __forceinline__ void tdm_load_2d_bf16(uint32 lds_addr, const void* gptr,
                                                 uint32 tile_x, uint32 tile_y,
                                                 uint32 stride_elems,
                                                 uint32 dim_x, uint32 dim_y) {
  unsigned long long ga = (unsigned long long)(size_t)gptr;
  uint4v g0;
  g0[0] = 1u;                                             // count=1, user mode
  g0[1] = lds_addr;                                       // lds_addr (bytes)
  g0[2] = (uint32)(ga & 0xffffffffu);                     // global_addr[31:0]
  g0[3] = (uint32)((ga >> 32) & 0x1ffffffu) | (2u << 30); // addr[56:32] | type=2
  int8v g1;
  g1[0] = (int)(1u << 16);                                // data_size=1 (2 bytes)
  g1[1] = (int)((dim_x & 0xffffu) << 16);                 // tensor_dim0[15:0]
  g1[2] = (int)((dim_x >> 16) | ((dim_y & 0xffffu) << 16));
  g1[3] = (int)((dim_y >> 16) | (tile_x << 16));          // tile_dim0
  g1[4] = (int)(tile_y & 0xffffu);                        // tile_dim1, tile_dim2=0
  g1[5] = (int)stride_elems;                              // tensor_dim0_stride lo
  g1[6] = 0;
  g1[7] = 0;
  int4v z4 = {0, 0, 0, 0};
  int8v z8 = {0, 0, 0, 0, 0, 0, 0, 0};
  __builtin_amdgcn_tensor_load_to_lds(g0, g1, z4, z4, z8, 0);
}

// ---------------------------------------------------------------------------
// K0: zero the grid-sync counters (runs first every launch -> deterministic)
// ---------------------------------------------------------------------------
__global__ void zero_counters(unsigned* cnt) {
  if (threadIdx.x < 2) cnt[threadIdx.x] = 0u;
}

// ---------------------------------------------------------------------------
// K1: f32 -> bf16 conversion (weights)
// ---------------------------------------------------------------------------
__global__ void cvt_f32_bf16(const float* __restrict__ src, bf16_t* __restrict__ dst, int n) {
  int i = blockIdx.x * 256 + threadIdx.x;
  if (i < n) dst[i] = (bf16_t)src[i];
}

// ---------------------------------------------------------------------------
// K2: embedding gather + bf16 convert (forward order and reversed order)
// ---------------------------------------------------------------------------
__global__ void gather_embed(const int* __restrict__ sent, const float* __restrict__ embed,
                             bf16_t* __restrict__ emb, bf16_t* __restrict__ embrev) {
  int t = blockIdx.x;
  int idx = sent[t];
  const float* row = embed + (size_t)idx * E_DIM;
  for (int k = threadIdx.x; k < E_DIM; k += blockDim.x) {
    bf16_t v = (bf16_t)row[k];
    emb[(size_t)t * E_DIM + k] = v;
    embrev[(size_t)(T_LEN - 1 - t) * E_DIM + k] = v;
  }
}

// ---------------------------------------------------------------------------
// K3: pre = emb @ w_ih^T + b_ih   (both directions, M=T, N=2048, K=1024)
// Block: 128 threads (4 waves). Tile: M=16 x N=128, K staged by 64.
// Double-buffered TDM pipeline: stage s+1's DMA is issued before stage s's
// math; s_wait_tensorcnt(2) (in-order TDM completion per wave) guarantees
// stage s has landed while s+1 is still in flight. Each wave runs 4 WMMAs
// per barrier pair (2 N-subtiles x 2 K-subchunks).
// LDS: (2x2KB A) + (2x16KB B) = 36 KB of the 320 KB WGP budget.
// ---------------------------------------------------------------------------
__global__ void gemm_pre(const bf16_t* __restrict__ embF, const bf16_t* __restrict__ embR,
                         const bf16_t* __restrict__ wihF, const bf16_t* __restrict__ wihB,
                         const float* __restrict__ bihF, const float* __restrict__ bihB,
                         float* __restrict__ preF, float* __restrict__ preB) {
  const int dir = blockIdx.z;
  const bf16_t* A  = dir ? embR : embF;
  const bf16_t* W  = dir ? wihB : wihF;
  const float*  bi = dir ? bihB : bihF;
  float*        out = dir ? preB : preF;

  const int t0 = blockIdx.x * 16;
  const int n0 = blockIdx.y * 128;

  __shared__ __align__(128) bf16_t Alds[2][16 * 64];
  __shared__ __align__(128) bf16_t Blds[2][128 * 64];

  const int lane = threadIdx.x & 31;
  const int wv   = threadIdx.x >> 5;
  const int row  = lane & 15;
  const int hsel = lane >> 4;

  v8f acc0 = {0.f, 0.f, 0.f, 0.f, 0.f, 0.f, 0.f, 0.f};
  v8f acc1 = {0.f, 0.f, 0.f, 0.f, 0.f, 0.f, 0.f, 0.f};

  const int S = E_DIM / 64;               // 16 K-stages

  // Prologue: DMA stage 0 into buffer 0.
  if (threadIdx.x == 0) {
    tdm_load_2d_bf16((uint32)(size_t)(void*)Alds[0],
                     A + (size_t)t0 * E_DIM, 64, 16, E_DIM, E_DIM, T_LEN);
    tdm_load_2d_bf16((uint32)(size_t)(void*)Blds[0],
                     W + (size_t)n0 * E_DIM, 64, 128, E_DIM, E_DIM, G4);
  }

  for (int s = 0; s < S; ++s) {
    const int buf = s & 1;
    if (threadIdx.x == 0) {
      if (s + 1 < S) {
        const int k1 = (s + 1) * 64;      // prefetch next stage into other buffer
        tdm_load_2d_bf16((uint32)(size_t)(void*)Alds[buf ^ 1],
                         A + (size_t)t0 * E_DIM + k1, 64, 16, E_DIM, E_DIM, T_LEN);
        tdm_load_2d_bf16((uint32)(size_t)(void*)Blds[buf ^ 1],
                         W + (size_t)n0 * E_DIM + k1, 64, 128, E_DIM, E_DIM, G4);
        __builtin_amdgcn_s_wait_tensorcnt(2);   // stage s complete, s+1 in flight
      } else {
        __builtin_amdgcn_s_wait_tensorcnt(0);   // drain last stage
      }
    }
    __syncthreads();                      // stage s tiles visible to all waves

    const bf16_t* Ab = Alds[buf];
    const bf16_t* Bb = Blds[buf];
#pragma unroll
    for (int ks = 0; ks < 2; ++ks) {
      v16bf a  = *(const v16bf*)&Ab[row * 64 + ks * 32 + hsel * 16];
      v16bf b0 = *(const v16bf*)&Bb[(wv * 32 + row) * 64 + ks * 32 + hsel * 16];
      v16bf b1 = *(const v16bf*)&Bb[(wv * 32 + 16 + row) * 64 + ks * 32 + hsel * 16];
      acc0 = __builtin_amdgcn_wmma_f32_16x16x32_bf16(false, a, false, b0,
                                                     (short)0, acc0, false, false);
      acc1 = __builtin_amdgcn_wmma_f32_16x16x32_bf16(false, a, false, b1,
                                                     (short)0, acc1, false, false);
    }
    __syncthreads();                      // done reading buf before it is re-DMA'd
  }

  const int na = n0 + wv * 32 + row;
  const int nb = na + 16;
  const float bva = bi[na];
  const float bvb = bi[nb];
#pragma unroll
  for (int v = 0; v < 8; ++v) {
    int m = v + hsel * 8;                 // C layout: lanes16-31 -> M+8
    out[(size_t)(t0 + m) * G4 + na] = acc0[v] + bva;
    out[(size_t)(t0 + m) * G4 + nb] = acc1[v] + bvb;
  }
}

// ---------------------------------------------------------------------------
// Device-scope arrive/spin barrier across the 16 blocks of one direction
// ---------------------------------------------------------------------------
__device__ __forceinline__ void grid_sync16(unsigned* cnt, unsigned target) {
  __threadfence();
  __syncthreads();
  if (threadIdx.x == 0) {
    __hip_atomic_fetch_add(cnt, 1u, __ATOMIC_RELEASE, __HIP_MEMORY_SCOPE_AGENT);
    while (__hip_atomic_load(cnt, __ATOMIC_ACQUIRE, __HIP_MEMORY_SCOPE_AGENT) < target) {
      __builtin_amdgcn_s_sleep(1);
    }
  }
  __syncthreads();
}

// ---------------------------------------------------------------------------
// K4: sequential LSTM recurrence, both directions.
// 32 blocks x 128 threads: dir = blk>>4, group = blk&15.
// Each block keeps its 128 w_hh gate rows (128 KB bf16) resident in LDS for
// all 4096 steps (CDNA5: 320 KB LDS per WGP). h double-buffered in global.
// ---------------------------------------------------------------------------
__global__ void lstm_recurrent(const float* __restrict__ preF, const float* __restrict__ preB,
                               const bf16_t* __restrict__ whhF, const bf16_t* __restrict__ whhB,
                               const float* __restrict__ bhhF, const float* __restrict__ bhhB,
                               const float* __restrict__ h0, const float* __restrict__ c0,
                               float* __restrict__ hs, float* __restrict__ hbuf,
                               unsigned* __restrict__ cnt) {
  const int dir = blockIdx.x >> 4;
  const int grp = blockIdx.x & 15;
  const int tid = threadIdx.x;            // 0..127
  const float*  pre = dir ? preB : preF;
  const bf16_t* whh = dir ? whhB : whhF;
  const float*  bhh = dir ? bhhB : bhhF;
  unsigned* mycnt = cnt + dir;
  float* hb = hbuf + dir * (2 * HH);      // two 512-float buffers

  extern __shared__ bf16_t wlds[];        // 128 rows * 512 = 128 KB
  __shared__ float hsh[HH];
  __shared__ float gsh[128];
  __shared__ float csh[32];

  const int gate = tid >> 5;              // i,f,g,o
  const int u    = tid & 31;
  const int grow = gate * HH + grp * 32 + u;   // global gate row

  // Stage this block's slice of w_hh into LDS (read from HBM exactly once).
  for (int k = 0; k < HH; ++k)
    wlds[tid * HH + k] = whh[(size_t)grow * HH + k];
  if (tid < 32) {
    csh[tid] = c0[dir * HH + grp * 32 + tid];
    hb[grp * 32 + tid] = h0[dir * HH + grp * 32 + tid];   // buffer 0 = h0
  }
  const float bpre = bhh[grow];

  unsigned epoch = 1;
  grid_sync16(mycnt, epoch * 16);         // weights + h0 visible everywhere

  for (int t = 0; t < T_LEN; ++t) {
    // broadcast h_t into LDS
    const float* hsrc = hb + (t & 1) * HH;
    for (int k = tid; k < HH; k += 128) hsh[k] = hsrc[k];
    __syncthreads();

    // gate pre-activation: pre (has b_ih) + b_hh + w_hh-row . h
    float acc = pre[(size_t)t * G4 + grow] + bpre;
    const bf16_t* wr = wlds + tid * HH;
#pragma unroll 8
    for (int k = 0; k < HH; ++k) acc += (float)wr[k] * hsh[k];
    gsh[tid] = acc;
    __syncthreads();

    if (tid < 32) {
      float ig = 1.f / (1.f + __expf(-gsh[tid]));
      float fg = 1.f / (1.f + __expf(-gsh[32 + tid]));
      float gg = tanhf(gsh[64 + tid]);
      float og = 1.f / (1.f + __expf(-gsh[96 + tid]));
      float c  = fg * csh[tid] + ig * gg;
      csh[tid] = c;
      float h  = og * tanhf(c);
      hb[((t + 1) & 1) * HH + grp * 32 + tid] = h;
      size_t trow = dir ? (size_t)(T_LEN - 1 - t) * (2 * HH) + HH
                        : (size_t)t * (2 * HH);
      hs[trow + grp * 32 + tid] = h;
    }
    ++epoch;
    grid_sync16(mycnt, epoch * 16);
  }
}

// ---------------------------------------------------------------------------
// K5: feats = hs @ w_out^T + b_out.  One wave per 16 timesteps; single
// 16(M=time) x 16(N=tag) WMMA tile, K=1024 in bf16 chunks of 32.
// ---------------------------------------------------------------------------
__global__ void feats_gemm(const float* __restrict__ hs, const float* __restrict__ w_out,
                           const float* __restrict__ b_out, float* __restrict__ feats) {
  const int blk  = blockIdx.x;            // 256 blocks
  const int lane = threadIdx.x & 31;
  const int row  = lane & 15;
  const int hsel = lane >> 4;

  v8f acc = {0.f, 0.f, 0.f, 0.f, 0.f, 0.f, 0.f, 0.f};
  for (int k0 = 0; k0 < 2 * HH; k0 += 32) {
    union { v16bf v; bf16_t e[16]; } a, b;
    const float* ap = hs    + (size_t)(blk * 16 + row) * (2 * HH) + k0 + hsel * 16;
    const float* bp = w_out + (size_t)row * (2 * HH)              + k0 + hsel * 16;
#pragma unroll
    for (int i = 0; i < 16; ++i) { a.e[i] = (bf16_t)ap[i]; b.e[i] = (bf16_t)bp[i]; }
    acc = __builtin_amdgcn_wmma_f32_16x16x32_bf16(false, a.v, false, b.v,
                                                  (short)0, acc, false, false);
  }
  const int n = row;
  const float bv = b_out[n];
#pragma unroll
  for (int v = 0; v < 8; ++v) {
    int m = v + hsel * 8;
    feats[(size_t)(blk * 16 + m) * KTAG + n] = acc[v] + bv;
  }
}

// ---------------------------------------------------------------------------
// K6: Viterbi decode (K=16, lane-per-tag) + backtrack.
// d_out[0] = path_score; d_out[1..T] = best path (as float).
// ---------------------------------------------------------------------------
__global__ void viterbi_kernel(const float* __restrict__ feats, const float* __restrict__ trans,
                               float* __restrict__ out, int* __restrict__ bptr, int out_size) {
  __shared__ float tr[KTAG * KTAG];
  __shared__ float prev[KTAG];
  __shared__ float nxt[KTAG];
  const int tid = threadIdx.x;            // 32 threads, 16 active
  for (int i = tid; i < KTAG * KTAG; i += 32) tr[i] = trans[i];
  if (tid < KTAG) prev[tid] = (tid == START_TAG) ? 0.f : NEG_F;
  __syncthreads();

  for (int t = 0; t < T_LEN; ++t) {
    if (tid < KTAG) {
      float best = prev[0] + tr[tid * KTAG + 0];
      int bi = 0;
#pragma unroll
      for (int p = 1; p < KTAG; ++p) {
        float s = prev[p] + tr[tid * KTAG + p];
        if (s > best) { best = s; bi = p; }
      }
      bptr[t * KTAG + tid] = bi;
      nxt[tid] = best + feats[(size_t)t * KTAG + tid];
    }
    __syncthreads();
    if (tid < KTAG) prev[tid] = nxt[tid];
    __syncthreads();
  }

  if (tid == 0) {
    float best = prev[0] + tr[STOP_TAG * KTAG + 0];
    int bi = 0;
    for (int p = 1; p < KTAG; ++p) {
      float s = prev[p] + tr[STOP_TAG * KTAG + p];
      if (s > best) { best = s; bi = p; }
    }
    out[0] = best;
    int tag = bi;
    for (int t = T_LEN - 1; t >= 0; --t) {
      if (1 + t < out_size) out[1 + t] = (float)tag;
      tag = bptr[t * KTAG + tag];
    }
  }
}

// ---------------------------------------------------------------------------
// Host-side launch
// ---------------------------------------------------------------------------
extern "C" void kernel_launch(void* const* d_in, const int* in_sizes, int n_in,
                              void* d_out, int out_size, void* d_ws, size_t ws_size,
                              hipStream_t stream) {
  (void)in_sizes; (void)n_in; (void)ws_size;
  const int*   sentence = (const int*)  d_in[0];
  const float* embed    = (const float*)d_in[1];
  const float* w_ih_f   = (const float*)d_in[2];
  const float* w_hh_f   = (const float*)d_in[3];
  const float* b_ih_f   = (const float*)d_in[4];
  const float* b_hh_f   = (const float*)d_in[5];
  const float* w_ih_b   = (const float*)d_in[6];
  const float* w_hh_b   = (const float*)d_in[7];
  const float* b_ih_b   = (const float*)d_in[8];
  const float* b_hh_b   = (const float*)d_in[9];
  const float* h0       = (const float*)d_in[10];
  const float* c0       = (const float*)d_in[11];
  const float* w_out    = (const float*)d_in[12];
  const float* b_out    = (const float*)d_in[13];
  const float* trans    = (const float*)d_in[14];
  float* out = (float*)d_out;

  char* ws = (char*)d_ws;
  size_t o = 0;
  bf16_t* emb    = (bf16_t*)(ws + o); o += (size_t)T_LEN * E_DIM * 2;   // 8 MB
  bf16_t* embrev = (bf16_t*)(ws + o); o += (size_t)T_LEN * E_DIM * 2;   // 8 MB
  bf16_t* wihf   = (bf16_t*)(ws + o); o += (size_t)G4 * E_DIM * 2;      // 4 MB
  bf16_t* wihb   = (bf16_t*)(ws + o); o += (size_t)G4 * E_DIM * 2;      // 4 MB
  bf16_t* whhf   = (bf16_t*)(ws + o); o += (size_t)G4 * HH * 2;         // 2 MB
  bf16_t* whhb   = (bf16_t*)(ws + o); o += (size_t)G4 * HH * 2;         // 2 MB
  float*  pref   = (float*) (ws + o); o += (size_t)T_LEN * G4 * 4;      // 32 MB
  float*  preb   = (float*) (ws + o); o += (size_t)T_LEN * G4 * 4;      // 32 MB
  float*  hs     = (float*) (ws + o); o += (size_t)T_LEN * 2 * HH * 4;  // 16 MB
  float*  feats  = (float*) (ws + o); o += (size_t)T_LEN * KTAG * 4;
  int*    bptr   = (int*)   (ws + o); o += (size_t)T_LEN * KTAG * 4;
  float*  hbuf   = (float*) (ws + o); o += 2 * 2 * HH * 4;
  unsigned* cnt  = (unsigned*)(ws + o); o += 64;

  zero_counters<<<1, 32, 0, stream>>>(cnt);

  const int n_ih = G4 * E_DIM, n_hh = G4 * HH;
  cvt_f32_bf16<<<(n_ih + 255) / 256, 256, 0, stream>>>(w_ih_f, wihf, n_ih);
  cvt_f32_bf16<<<(n_ih + 255) / 256, 256, 0, stream>>>(w_ih_b, wihb, n_ih);
  cvt_f32_bf16<<<(n_hh + 255) / 256, 256, 0, stream>>>(w_hh_f, whhf, n_hh);
  cvt_f32_bf16<<<(n_hh + 255) / 256, 256, 0, stream>>>(w_hh_b, whhb, n_hh);

  gather_embed<<<T_LEN, 256, 0, stream>>>(sentence, embed, emb, embrev);

  dim3 ggrid(T_LEN / 16, G4 / 128, 2);
  gemm_pre<<<ggrid, 128, 0, stream>>>(emb, embrev, wihf, wihb, b_ih_f, b_ih_b, pref, preb);

  lstm_recurrent<<<32, 128, 128 * HH * sizeof(bf16_t), stream>>>(
      pref, preb, whhf, whhb, b_hh_f, b_hh_b, h0, c0, hs, hbuf, cnt);

  feats_gemm<<<T_LEN / 16, 32, 0, stream>>>(hs, w_out, b_out, feats);

  viterbi_kernel<<<1, 32, 0, stream>>>(feats, trans, out, bptr, out_size);
}
